// MHSAModule_58394375356654
// MI455X (gfx1250) — compile-verified, matrix operand order checked by reading, not run on hardware
//
#include <hip/hip_runtime.h>
#include <hip/hip_bf16.h>

typedef __attribute__((ext_vector_type(16))) _Float16 v16h;
typedef __attribute__((ext_vector_type(8)))  _Float16 v8h;
typedef __attribute__((ext_vector_type(8)))  float    v8f;

#define DIM    1024
#define HEADS  16
#define SEQ    2048
#define HDIM   64
#define TOKENS 8192   // B*S = 4*2048

// ---------------------------------------------------------------------------
// A/B fragment loader for v_wmma_f32_16x16x32_f16.
// A (16x32, M x K): lane 0-15 = row M, lanes L and L+16 share row M=L.
//   halves 0..7  = K {base..base+7},  halves 8..15 = K {base+16..base+23},
//   base = 8*(lane>=16).
// B (32x16, K x N) mirrors this with lane = N column, so for B = W^T the
// fragment is loaded from *rows* of W with the identical pattern.
// ---------------------------------------------------------------------------
__device__ __forceinline__ v16h load_frag(const _Float16* base, int ld, int lane) {
    int row = lane & 15;
    int hi  = (lane >> 4) & 1;
    const _Float16* p = base + row * ld + hi * 8;
    v8h lo  = *(const v8h*)(p);       // K base .. base+7
    v8h hiv = *(const v8h*)(p + 16);  // K base+16 .. base+23
    return __builtin_shufflevector(lo, hiv, 0,1,2,3,4,5,6,7,8,9,10,11,12,13,14,15);
}

__device__ __forceinline__ v8f wmma16(v16h a, v16h b, v8f c) {
    return __builtin_amdgcn_wmma_f32_16x16x32_f16(
        false, a, false, b, (short)0, c, false, false);
}

// Fragment set for a 32x64 GEMM macro-tile step (K=32)
struct Frags { v16h a0, a1, b0, b1, b2, b3; };

__device__ __forceinline__ void load_set(Frags& f,
                                         const _Float16* xr0, const _Float16* xr1,
                                         const _Float16* wr, int k, int lane) {
    f.a0 = load_frag(xr0 + k, DIM, lane);
    f.a1 = load_frag(xr1 + k, DIM, lane);
    f.b0 = load_frag(wr + 0  * DIM + k, DIM, lane);
    f.b1 = load_frag(wr + 16 * DIM + k, DIM, lane);
    f.b2 = load_frag(wr + 32 * DIM + k, DIM, lane);
    f.b3 = load_frag(wr + 48 * DIM + k, DIM, lane);
}

__device__ __forceinline__ void mma_set(v8f acc[2][4], const Frags& f) {
    acc[0][0] = wmma16(f.a0, f.b0, acc[0][0]);
    acc[0][1] = wmma16(f.a0, f.b1, acc[0][1]);
    acc[0][2] = wmma16(f.a0, f.b2, acc[0][2]);
    acc[0][3] = wmma16(f.a0, f.b3, acc[0][3]);
    acc[1][0] = wmma16(f.a1, f.b0, acc[1][0]);
    acc[1][1] = wmma16(f.a1, f.b1, acc[1][1]);
    acc[1][2] = wmma16(f.a1, f.b2, acc[1][2]);
    acc[1][3] = wmma16(f.a1, f.b3, acc[1][3]);
}

// ---------------------------------------------------------------------------
// Kernel 1: f32 -> f16 conversion, 8 elements/thread (b128 in, b128 out)
// ---------------------------------------------------------------------------
__global__ void cvt_f32_f16_v8(const float* __restrict__ src,
                               _Float16* __restrict__ dst, int n) {
    int i = (blockIdx.x * blockDim.x + threadIdx.x) * 8;
    if (i >= n) return;
    float4 f0 = *(const float4*)(src + i);
    float4 f1 = *(const float4*)(src + i + 4);
    v8h h;
    h[0] = (_Float16)f0.x; h[1] = (_Float16)f0.y;
    h[2] = (_Float16)f0.z; h[3] = (_Float16)f0.w;
    h[4] = (_Float16)f1.x; h[5] = (_Float16)f1.y;
    h[6] = (_Float16)f1.z; h[7] = (_Float16)f1.w;
    *(v8h*)(dst + i) = h;
}

// ---------------------------------------------------------------------------
// Kernel 2: fused QKV projection.  Y = x @ W.T + b for W in {Wq,Wk,Wv}.
// One wave computes a 32(token) x 64(feature) tile; double-buffered K loop,
// FULLY UNROLLED (no backedge -> no phi copies, all-immediate addressing).
// Q,K -> [B,H,S,HD] f16.  V -> transposed [B,H,HD,S] f16.
// ---------------------------------------------------------------------------
__global__ __launch_bounds__(256) void qkv_gemm(
    const _Float16* __restrict__ xh,
    const _Float16* __restrict__ wqh, const _Float16* __restrict__ wkh,
    const _Float16* __restrict__ wvh,
    const float* __restrict__ bq, const float* __restrict__ bk,
    const float* __restrict__ bv,
    _Float16* __restrict__ qh, _Float16* __restrict__ kh,
    _Float16* __restrict__ vt) {
    const int lane = threadIdx.x & 31;
    const int wave = threadIdx.x >> 5;
    const int NT = (3 * DIM) / 64;                 // 48 n-tiles of 64
    int tile = blockIdx.x * 8 + wave;              // 256*48 = 12288 tiles
    int mt = tile / NT;
    int nt = tile % NT;
    int row0 = mt * 32;
    int col0 = nt * 64;
    int mat  = col0 >> 10;                         // 0=q 1=k 2=v
    int f0   = col0 & (DIM - 1);
    const _Float16* W    = (mat == 0) ? wqh : (mat == 1) ? wkh : wvh;
    const float*    bias = (mat == 0) ? bq  : (mat == 1) ? bk  : bv;

    const _Float16* xr0 = xh + (size_t)row0 * DIM;
    const _Float16* xr1 = xh + (size_t)(row0 + 16) * DIM;
    const _Float16* wr  = W  + (size_t)f0 * DIM;

    v8f acc[2][4] = {};
    Frags fA, fB;
    load_set(fA, xr0, xr1, wr, 0, lane);
    #pragma unroll
    for (int k0 = 0; k0 < DIM - 64; k0 += 64) {
        load_set(fB, xr0, xr1, wr, k0 + 32, lane);   // in flight during fA WMMAs
        mma_set(acc, fA);
        load_set(fA, xr0, xr1, wr, k0 + 64, lane);   // in flight during fB WMMAs
        mma_set(acc, fB);
    }
    load_set(fB, xr0, xr1, wr, DIM - 32, lane);
    mma_set(acc, fA);
    mma_set(acc, fB);

    // C layout: lane = column, vgpr r = row r (+8 for lanes 16-31)
    int colLane = lane & 15;
    int half    = lane >> 4;
    #pragma unroll
    for (int m = 0; m < 2; ++m) {
        #pragma unroll
        for (int t = 0; t < 4; ++t) {
            int f = f0 + t * 16 + colLane;
            float bval = bias[f];
            int h = f >> 6, d = f & 63;
            #pragma unroll
            for (int r = 0; r < 8; ++r) {
                int tok = row0 + m * 16 + r + half * 8;
                int b_  = tok >> 11;
                int s   = tok & (SEQ - 1);
                _Float16 hv = (_Float16)(acc[m][t][r] + bval);
                if (mat == 2)
                    vt[(((size_t)(b_ * HEADS + h) * HDIM + d) * SEQ) + s] = hv;
                else if (mat == 0)
                    qh[(((size_t)(b_ * HEADS + h) * SEQ + s) * HDIM) + d] = hv;
                else
                    kh[(((size_t)(b_ * HEADS + h) * SEQ + s) * HDIM) + d] = hv;
            }
        }
    }
}

// ---------------------------------------------------------------------------
// Kernel 3: flash-style attention.  One wave = one 32-query block of one
// (batch,head).  Streams keys 32 at a time; online softmax; V fragments are
// loaded right after the score WMMAs so the softmax VALU hides their latency;
// each V fragment feeds two PV WMMAs.  Next block prefetched via
// global_prefetch_b8.
// ---------------------------------------------------------------------------
__global__ __launch_bounds__(256) void attn_kernel(
    const _Float16* __restrict__ qh, const _Float16* __restrict__ kh,
    const _Float16* __restrict__ vt, _Float16* __restrict__ attn_h) {
    __shared__ __align__(16) _Float16 pbuf[8][2][16 * 32]; // per-wave P staging
    const int lane = threadIdx.x & 31;
    const int wave = threadIdx.x >> 5;
    int idx = blockIdx.x * 8 + wave;        // B*H*(S/32) = 4096 waves
    int qt  = idx & 63;                     // 32-query tile within sequence
    int bh  = idx >> 6;                     // 0..63 = b*16+h

    const _Float16* Q = qh + (size_t)bh * SEQ * HDIM + (size_t)qt * 32 * HDIM;
    const _Float16* K = kh + (size_t)bh * SEQ * HDIM;
    const _Float16* V = vt + (size_t)bh * HDIM * SEQ;

    v16h qa[2][2];
    qa[0][0] = load_frag(Q + 0,              HDIM, lane);
    qa[0][1] = load_frag(Q + 32,             HDIM, lane);
    qa[1][0] = load_frag(Q + 16 * HDIM,      HDIM, lane);
    qa[1][1] = load_frag(Q + 16 * HDIM + 32, HDIM, lane);

    v8f  o[2][4] = {};
    float mrow[2][8], lrow[2][8];
    #pragma unroll
    for (int m = 0; m < 2; ++m)
        #pragma unroll
        for (int r = 0; r < 8; ++r) { mrow[m][r] = -3.0e38f; lrow[m][r] = 0.0f; }
    const float scale  = 0.125f;            // 1/sqrt(64)
    const int colLane  = lane & 15;
    const int half     = lane >> 4;

    for (int j0 = 0; j0 < SEQ; j0 += 32) {
        // ---- prefetch next key/value block while this one computes ----
        if (j0 + 32 < SEQ) {
            __builtin_prefetch(K + (size_t)(j0 + 32 + lane) * HDIM, 0, 0);
            __builtin_prefetch(V + (size_t)lane * SEQ + j0 + 32, 0, 0);
            __builtin_prefetch(V + (size_t)(32 + lane) * SEQ + j0 + 32, 0, 0);
        }
        // ---- scores: 2 key tiles x 2 query sub-tiles ----
        v8f sc[2][2];
        #pragma unroll
        for (int t = 0; t < 2; ++t) {
            const _Float16* Kt = K + (size_t)(j0 + t * 16) * HDIM;
            v16h kb0 = load_frag(Kt + 0,  HDIM, lane);
            v16h kb1 = load_frag(Kt + 32, HDIM, lane);
            #pragma unroll
            for (int m = 0; m < 2; ++m) {
                v8f c = {};
                c = wmma16(qa[m][0], kb0, c);
                c = wmma16(qa[m][1], kb1, c);
                #pragma unroll
                for (int r = 0; r < 8; ++r) c[r] *= scale;
                sc[m][t] = c;
            }
        }
        // ---- issue V fragment loads now; softmax below hides their latency ----
        v16h vb[4];
        #pragma unroll
        for (int t = 0; t < 4; ++t)
            vb[t] = load_frag(V + (size_t)(t * 16) * SEQ + j0, SEQ, lane);

        // ---- online softmax + stage P in LDS ----
        #pragma unroll
        for (int m = 0; m < 2; ++m) {
            _Float16* pb = &pbuf[wave][m][0];
            #pragma unroll
            for (int r = 0; r < 8; ++r) {
                float mx = fmaxf(sc[m][0][r], sc[m][1][r]);
                #pragma unroll
                for (int off = 1; off < 16; off <<= 1)
                    mx = fmaxf(mx, __shfl_xor(mx, off, 32));
                float mo = mrow[m][r];
                float mn = fmaxf(mo, mx);
                float corr = __expf(mo - mn);
                mrow[m][r] = mn;
                lrow[m][r] *= corr;
                #pragma unroll
                for (int t = 0; t < 4; ++t) o[m][t][r] *= corr;
                float e0 = __expf(sc[m][0][r] - mn);
                float e1 = __expf(sc[m][1][r] - mn);
                int mr = r + 8 * half;
                pb[mr * 32 + 0  + colLane] = (_Float16)e0;
                pb[mr * 32 + 16 + colLane] = (_Float16)e1;
                float ps = e0 + e1;
                #pragma unroll
                for (int off = 1; off < 16; off <<= 1)
                    ps += __shfl_xor(ps, off, 32);
                lrow[m][r] += ps;
            }
        }
        // ---- P x V: each V fragment feeds both query sub-tiles ----
        v16h pa0 = load_frag(&pbuf[wave][0][0], 32, lane);
        v16h pa1 = load_frag(&pbuf[wave][1][0], 32, lane);
        #pragma unroll
        for (int t = 0; t < 4; ++t) {
            o[0][t] = wmma16(pa0, vb[t], o[0][t]);
            o[1][t] = wmma16(pa1, vb[t], o[1][t]);
        }
    }

    // ---- normalize and write out: attn_h[token][h*64 + d] ----
    int b_ = bh >> 4, h = bh & 15;
    #pragma unroll
    for (int m = 0; m < 2; ++m) {
        #pragma unroll
        for (int r = 0; r < 8; ++r) {
            float inv = 1.0f / lrow[m][r];
            int s   = qt * 32 + m * 16 + r + 8 * half;
            int tok = b_ * SEQ + s;
            #pragma unroll
            for (int t = 0; t < 4; ++t) {
                int col = h * HDIM + t * 16 + colLane;
                attn_h[(size_t)tok * DIM + col] = (_Float16)(o[m][t][r] * inv);
            }
        }
    }
}

// ---------------------------------------------------------------------------
// Kernel 4: output projection.  out = attn @ Wo.T + bo (f32), 32x64 tiles,
// double-buffered + fully unrolled like qkv_gemm.
// ---------------------------------------------------------------------------
__global__ __launch_bounds__(256) void out_gemm(
    const _Float16* __restrict__ ah, const _Float16* __restrict__ woh,
    const float* __restrict__ bo, float* __restrict__ out) {
    const int lane = threadIdx.x & 31;
    const int wave = threadIdx.x >> 5;
    const int NT = DIM / 64;                       // 16 n-tiles
    int tile = blockIdx.x * 8 + wave;              // 256*16 = 4096 tiles
    int mt = tile / NT;
    int nt = tile % NT;
    int row0 = mt * 32;
    int f0   = nt * 64;

    const _Float16* ar0 = ah  + (size_t)row0 * DIM;
    const _Float16* ar1 = ah  + (size_t)(row0 + 16) * DIM;
    const _Float16* wr  = woh + (size_t)f0 * DIM;

    v8f acc[2][4] = {};
    Frags fA, fB;
    load_set(fA, ar0, ar1, wr, 0, lane);
    #pragma unroll
    for (int k0 = 0; k0 < DIM - 64; k0 += 64) {
        load_set(fB, ar0, ar1, wr, k0 + 32, lane);
        mma_set(acc, fA);
        load_set(fA, ar0, ar1, wr, k0 + 64, lane);
        mma_set(acc, fB);
    }
    load_set(fB, ar0, ar1, wr, DIM - 32, lane);
    mma_set(acc, fA);
    mma_set(acc, fB);

    int colLane = lane & 15;
    int half    = lane >> 4;
    #pragma unroll
    for (int m = 0; m < 2; ++m) {
        #pragma unroll
        for (int t = 0; t < 4; ++t) {
            int f = f0 + t * 16 + colLane;
            float bval = bo[f];
            #pragma unroll
            for (int r = 0; r < 8; ++r) {
                int tok = row0 + m * 16 + r + half * 8;
                out[(size_t)tok * DIM + f] = acc[m][t][r] + bval;
            }
        }
    }
}

// ---------------------------------------------------------------------------
extern "C" void kernel_launch(void* const* d_in, const int* in_sizes, int n_in,
                              void* d_out, int out_size, void* d_ws, size_t ws_size,
                              hipStream_t stream) {
    const float* x  = (const float*)d_in[0];
    const float* Wq = (const float*)d_in[1];
    const float* bq = (const float*)d_in[2];
    const float* Wk = (const float*)d_in[3];
    const float* bk = (const float*)d_in[4];
    const float* Wv = (const float*)d_in[5];
    const float* bv = (const float*)d_in[6];
    const float* Wo = (const float*)d_in[7];
    const float* bo = (const float*)d_in[8];
    float* out = (float*)d_out;

    char* ws = (char*)d_ws;
    const size_t XH_BYTES = (size_t)TOKENS * DIM * 2;   // 16 MB
    const size_t W_BYTES  = (size_t)DIM * DIM * 2;      //  2 MB
    _Float16* xh  = (_Float16*)(ws);
    _Float16* wqh = (_Float16*)(ws + XH_BYTES);
    _Float16* wkh = (_Float16*)(ws + XH_BYTES + W_BYTES);
    _Float16* wvh = (_Float16*)(ws + XH_BYTES + 2 * W_BYTES);
    _Float16* woh = (_Float16*)(ws + XH_BYTES + 3 * W_BYTES);
    _Float16* qh  = (_Float16*)(ws + XH_BYTES + 4 * W_BYTES);
    _Float16* kh  = (_Float16*)(ws + 2 * XH_BYTES + 4 * W_BYTES);
    _Float16* vt  = (_Float16*)(ws + 3 * XH_BYTES + 4 * W_BYTES);
    _Float16* attn_h = xh;   // xh dead after qkv_gemm; reuse for attention out

    int nx = TOKENS * DIM;   // 8388608
    int nw = DIM * DIM;      // 1048576
    cvt_f32_f16_v8<<<nx / (256 * 8), 256, 0, stream>>>(x,  xh,  nx);
    cvt_f32_f16_v8<<<nw / (256 * 8), 256, 0, stream>>>(Wq, wqh, nw);
    cvt_f32_f16_v8<<<nw / (256 * 8), 256, 0, stream>>>(Wk, wkh, nw);
    cvt_f32_f16_v8<<<nw / (256 * 8), 256, 0, stream>>>(Wv, wvh, nw);
    cvt_f32_f16_v8<<<nw / (256 * 8), 256, 0, stream>>>(Wo, woh, nw);

    // QKV: (TOKENS/32) * (3*DIM/64) = 256*48 = 12288 waves, 8 per block
    qkv_gemm<<<12288 / 8, 256, 0, stream>>>(xh, wqh, wkh, wvh, bq, bk, bv,
                                            qh, kh, vt);
    // attention: B*H*(S/32) = 4096 waves, 8 per block
    attn_kernel<<<4096 / 8, 256, 0, stream>>>(qh, kh, vt, attn_h);
    // output projection: (TOKENS/32) * (DIM/64) = 4096 waves, 8 per block
    out_gemm<<<4096 / 8, 256, 0, stream>>>(attn_h, woh, bo, out);
}
